// model_8959301779746
// MI455X (gfx1250) — matrix operand order repaired and checked by reading, not compiled, not run on hardware
//
#include <hip/hip_runtime.h>
#include <hip/hip_bf16.h>

// CDNA5 / gfx1250 GCN forward.
// - Dense transforms: v_wmma_f32_16x16x32_bf16, wave32. One wave computes a
//   full 16 x Mo row strip (Mo/16 accumulator tiles), loading each A fragment
//   once per k-step and reusing it across all column tiles. W pre-swizzled into
//   B-fragment order (contiguous 32B per lane) and L2/WGP$-resident.
//   Activations carried as bf16 (f32 accumulate).
// - Message passing: CSR-by-destination built once (int histogram + LDS scan +
//   cursor fill), then atomic-free gather with fused bias/ReLU/dropout epilogue.
//   Avoids ~425M f32 atomics that would cap below the 23.3 TB/s read roofline.

typedef __bf16 bf16_t;
typedef __attribute__((ext_vector_type(16))) __bf16 v16bf;
typedef __attribute__((ext_vector_type(8)))  __bf16 v8bf;
typedef __attribute__((ext_vector_type(8)))  float  v8f;

// ---------------- CSR / normalization setup ----------------
__global__ void k_zero_i32(int* __restrict__ p, int n) {
  int i = blockIdx.x * blockDim.x + threadIdx.x;
  if (i < n) p[i] = 0;
}

__global__ void k_count(const int* __restrict__ dst, int* __restrict__ count, int e) {
  int i = blockIdx.x * blockDim.x + threadIdx.x;
  if (i < e) atomicAdd(&count[dst[i]], 1);
}

// Single-block exclusive scan (Hillis-Steele in LDS) over count -> rowptr, cursor.
__global__ __launch_bounds__(1024)
void k_scan_rowptr(const int* __restrict__ count, int* __restrict__ rowptr,
                   int* __restrict__ cursor, int n) {
  __shared__ int sbuf[1024];
  __shared__ int s_off;
  const int tid = threadIdx.x;
  if (tid == 0) s_off = 0;
  __syncthreads();
  for (int base = 0; base < n; base += 1024) {
    int i = base + tid;
    int v = (i < n) ? count[i] : 0;
    sbuf[tid] = v;
    __syncthreads();
#pragma unroll
    for (int off = 1; off < 1024; off <<= 1) {
      int t = (tid >= off) ? sbuf[tid - off] : 0;
      __syncthreads();
      sbuf[tid] += t;
      __syncthreads();
    }
    int excl = sbuf[tid] - v;
    if (i < n) { rowptr[i] = s_off + excl; cursor[i] = s_off + excl; }
    __syncthreads();
    if (tid == 0) s_off += sbuf[1023];
    __syncthreads();
  }
  if (tid == 0) rowptr[n] = s_off;
}

__global__ void k_fill(const int* __restrict__ dst, int* __restrict__ cursor,
                       int* __restrict__ eidx, int e) {
  int i = blockIdx.x * blockDim.x + threadIdx.x;
  if (i >= e) return;
  int slot = atomicAdd(&cursor[dst[i]], 1);
  eidx[slot] = i;
}

__global__ void k_dinv_deg(const int* __restrict__ count, float* __restrict__ dinv, int n) {
  int i = blockIdx.x * blockDim.x + threadIdx.x;
  if (i < n) dinv[i] = rsqrtf((float)count[i] + 1.0f);   // +1 self-loop
}

__global__ void k_norm(const int* __restrict__ src, const int* __restrict__ dst,
                       const float* __restrict__ dinv, float* __restrict__ norm, int e) {
  int i = blockIdx.x * blockDim.x + threadIdx.x;
  if (i < e) norm[i] = dinv[src[i]] * dinv[dst[i]];
}

__global__ void k_cvt_bf16(const float* __restrict__ in, bf16_t* __restrict__ out, long long n) {
  long long i = (long long)blockIdx.x * blockDim.x + threadIdx.x;
  if (i < n) out[i] = (bf16_t)in[i];
}

// Pre-swizzle W [K x Mo] f32 row-major into k_gemm_wmma's B-fragment order (bf16, same k order as A):
// fragment (kt, ct): lane l holds 16 contiguous bf16, element e -> K = kt*32 + (l>>4)*16 + e,
// column = ct*16 + (l&15).  Linear index = ((kt*(Mo/16)+ct)*32 + lane)*16 + e.
__global__ void k_pack_w(const float* __restrict__ W, bf16_t* __restrict__ out, int K, int Mo) {
  int i = blockIdx.x * blockDim.x + threadIdx.x;
  if (i >= K * Mo) return;
  int k = i / Mo, m = i - k * Mo;
  int kt = k >> 5, r = k & 31;
  int lane = ((r >> 4) << 4) | (m & 15);
  int e = r & 15;
  int ct = m >> 4;
  long long idx = ((((long long)kt * (Mo >> 4) + ct) * 32) + lane) * 16 + e;
  out[idx] = (bf16_t)W[i];
}

// ---------------- WMMA GEMM: C[N x Mo] = A[N x K](bf16) * Wswz(bf16) ----------------
// One wave per 16-row strip; Mo/16 accumulator tiles, A fragment reused across
// column tiles. grid = N/16, blockDim = 32. K/32 * Mo/16 static v_wmma each.
template <int K, int Mo>
__global__ __launch_bounds__(32)
void k_gemm_wmma(const bf16_t* __restrict__ A, const bf16_t* __restrict__ Wswz,
                 float* __restrict__ C) {
  constexpr int CT = Mo / 16;
  const int lane = threadIdx.x;
  const int half = lane >> 4;
  const int ml   = lane & 15;

  const bf16_t* arow = A + ((long long)blockIdx.x * 16 + ml) * K + half * 16;
  const bf16_t* wlane = Wswz + lane * 16;

  v8f acc[CT];
#pragma unroll
  for (int ct = 0; ct < CT; ++ct) acc[ct] = (v8f){};

#pragma unroll
  for (int kt = 0; kt < K / 32; ++kt) {
    __builtin_prefetch(arow + kt * 32 + 64, 0, 1);   // global_prefetch_b8
    v8bf a0 = *(const v8bf*)(arow + kt * 32);
    v8bf a1 = *(const v8bf*)(arow + kt * 32 + 8);
    v16bf a = __builtin_shufflevector(a0, a1, 0, 1, 2, 3, 4, 5, 6, 7,
                                      8, 9, 10, 11, 12, 13, 14, 15);
#pragma unroll
    for (int ct = 0; ct < CT; ++ct) {
      v16bf b = *(const v16bf*)(wlane + (long long)(kt * CT + ct) * 512);
      acc[ct] = __builtin_amdgcn_wmma_f32_16x16x32_bf16(
          false, a, false, b, (short)0, acc[ct], false, false);
    }
  }

  float* cp = C + ((long long)blockIdx.x * 16 + half * 8) * Mo + ml;
#pragma unroll
  for (int ct = 0; ct < CT; ++ct)
#pragma unroll
    for (int v = 0; v < 8; ++v)
      cp[(long long)v * Mo + ct * 16] = acc[ct][v];
}

// ---------------- CSR gather + fused epilogue ----------------
// One wave per node; lane owns D/32 contiguous features. acc starts with the
// self-loop term, accumulates norm-weighted neighbor rows, then bias/ReLU/dropout.
template <int D>
__global__ void k_gather_post(const int* __restrict__ rowptr, const int* __restrict__ eidx,
                              const int* __restrict__ src, const float* __restrict__ norm,
                              const float* __restrict__ h, const float* __restrict__ dinv,
                              const float* __restrict__ bias, const float* __restrict__ u,
                              bf16_t* __restrict__ out_bf, float* __restrict__ out_f32,
                              int n, int do_drop) {
  constexpr int CH = D / 32;              // floats per lane (4 or 8)
  const int wave = (blockIdx.x * blockDim.x + threadIdx.x) >> 5;
  const int lane = threadIdx.x & 31;
  if (wave >= n) return;

  const int beg = rowptr[wave], end = rowptr[wave + 1];
  const float di = dinv[wave];
  const float* hrow = h + (long long)wave * D + lane * CH;

  float acc[CH];
#pragma unroll
  for (int c = 0; c < CH; ++c) acc[c] = di * di * hrow[c];   // self-loop

  for (int j = beg; j < end; ++j) {
    int e = eidx[j];
    int s = src[e];
    float nr = norm[e];
    const float* hs = h + (long long)s * D + lane * CH;
#pragma unroll
    for (int c = 0; c < CH; ++c) acc[c] = fmaf(nr, hs[c], acc[c]);
  }

  const long long obase = (long long)wave * D + lane * CH;
#pragma unroll
  for (int c = 0; c < CH; ++c) {
    float v = fmaxf(acc[c] + bias[lane * CH + c], 0.0f);
    if (do_drop) v = (u[obase + c] > 0.5f) ? v * 2.0f : 0.0f;
    if (out_bf)  out_bf[obase + c]  = (bf16_t)v;
    if (out_f32) out_f32[obase + c] = v;
  }
}

// small-d gather (d = 1 or 2), one thread per node, relu only
__global__ void k_gather_small(const int* __restrict__ rowptr, const int* __restrict__ eidx,
                               const int* __restrict__ src, const float* __restrict__ norm,
                               const float* __restrict__ h, const float* __restrict__ dinv,
                               const float* __restrict__ bias, float* __restrict__ out,
                               int n, int d) {
  int i = blockIdx.x * blockDim.x + threadIdx.x;
  if (i >= n) return;
  float di = dinv[i];
  float acc[2];
  for (int c = 0; c < d; ++c) acc[c] = di * di * h[(long long)i * d + c];
  int beg = rowptr[i], end = rowptr[i + 1];
  for (int j = beg; j < end; ++j) {
    int e = eidx[j];
    int s = src[e];
    float nr = norm[e];
    for (int c = 0; c < d; ++c) acc[c] = fmaf(nr, h[(long long)s * d + c], acc[c]);
  }
  for (int c = 0; c < d; ++c)
    out[(long long)i * d + c] = fmaxf(acc[c] + bias[c], 0.0f);
}

// tiny layer transforms
__global__ void k_gemm_d2(const bf16_t* __restrict__ x, const float* __restrict__ W,
                          float* __restrict__ out, int n) {
  int i = blockIdx.x * blockDim.x + threadIdx.x;
  if (i >= n) return;
  const bf16_t* xr = x + (long long)i * 128;
  float a0 = 0.f, a1 = 0.f;
#pragma unroll 8
  for (int k = 0; k < 128; ++k) {
    float xv = (float)xr[k];
    a0 = fmaf(xv, W[2 * k + 0], a0);
    a1 = fmaf(xv, W[2 * k + 1], a1);
  }
  out[2 * i + 0] = a0;
  out[2 * i + 1] = a1;
}

__global__ void k_gemm_d1(const float* __restrict__ x, const float* __restrict__ W,
                          float* __restrict__ out, int n) {
  int i = blockIdx.x * blockDim.x + threadIdx.x;
  if (i >= n) return;
  out[i] = fmaf(x[2 * i], W[0], x[2 * i + 1] * W[1]);
}

static inline int cdiv_ll(long long a, long long b) { return (int)((a + b - 1) / b); }

extern "C" void kernel_launch(void* const* d_in, const int* in_sizes, int n_in,
                              void* d_out, int out_size, void* d_ws, size_t ws_size,
                              hipStream_t stream) {
  (void)n_in; (void)out_size; (void)ws_size;
  const float* x  = (const float*)d_in[0];
  const int*   ei = (const int*)d_in[1];
  const float* u1 = (const float*)d_in[2];
  const float* u2 = (const float*)d_in[3];
  const float* u3 = (const float*)d_in[4];
  const float* W1 = (const float*)d_in[5];
  const float* b1 = (const float*)d_in[6];
  const float* W2 = (const float*)d_in[7];
  const float* b2 = (const float*)d_in[8];
  const float* W3 = (const float*)d_in[9];
  const float* b3 = (const float*)d_in[10];
  const float* W4 = (const float*)d_in[11];
  const float* b4 = (const float*)d_in[12];
  const float* W5 = (const float*)d_in[13];
  const float* b5 = (const float*)d_in[14];

  const int N = in_sizes[0] / 256;   // 50000 (divisible by 16)
  const int E = in_sizes[1] / 2;     // 800000
  const int* src = ei;
  const int* dst = ei + E;

  // workspace layout
  float*  dinv  = (float*)d_ws;                          // N f32
  float*  norm  = dinv + N;                              // E f32
  float*  bufA  = norm + E;                              // N*2 f32 used (small layers)
  float*  bufB  = bufA + (long long)N * 256;             // N*256 f32 (h = x@W)
  bf16_t* abf   = (bf16_t*)(bufB + (long long)N * 256);  // N*256 bf16 activations
  bf16_t* wswz1 = abf + (long long)N * 256;              // 256*128 bf16
  bf16_t* wswz2 = wswz1 + 256 * 128;                     // 128*256 bf16
  bf16_t* wswz3 = wswz2 + 128 * 256;                     // 256*128 bf16
  int*    count  = (int*)(wswz3 + 256 * 128);            // N
  int*    rowptr = count + N;                            // N+1
  int*    cursor = rowptr + N + 1;                       // N
  int*    eidx   = cursor + N;                           // E

  const int T = 256;
  const int RT = N / 16;

  // --- one-time prep: CSR, normalization, weight swizzle, x -> bf16 ---
  k_zero_i32   <<<cdiv_ll(N, T), T, 0, stream>>>(count, N);
  k_count      <<<cdiv_ll(E, T), T, 0, stream>>>(dst, count, E);
  k_scan_rowptr<<<1, 1024, 0, stream>>>(count, rowptr, cursor, N);
  k_fill       <<<cdiv_ll(E, T), T, 0, stream>>>(dst, cursor, eidx, E);
  k_dinv_deg   <<<cdiv_ll(N, T), T, 0, stream>>>(count, dinv, N);
  k_norm       <<<cdiv_ll(E, T), T, 0, stream>>>(src, dst, dinv, norm, E);
  k_pack_w     <<<cdiv_ll(256 * 128, T), T, 0, stream>>>(W1, wswz1, 256, 128);
  k_pack_w     <<<cdiv_ll(128 * 256, T), T, 0, stream>>>(W2, wswz2, 128, 256);
  k_pack_w     <<<cdiv_ll(256 * 128, T), T, 0, stream>>>(W3, wswz3, 256, 128);
  k_cvt_bf16   <<<cdiv_ll((long long)N * 256, T), T, 0, stream>>>(x, abf, (long long)N * 256);

  const int GW = cdiv_ll((long long)N * 32, T);   // wave-per-node gather grid

  // --- layer 1: 256 -> 128 ---
  k_gemm_wmma<256, 128><<<RT, 32, 0, stream>>>(abf, wswz1, bufB);
  k_gather_post<128><<<GW, T, 0, stream>>>(rowptr, eidx, src, norm, bufB, dinv,
                                           b1, u1, abf, nullptr, N, 1);
  // --- layer 2: 128 -> 256 ---
  k_gemm_wmma<128, 256><<<RT, 32, 0, stream>>>(abf, wswz2, bufB);
  k_gather_post<256><<<GW, T, 0, stream>>>(rowptr, eidx, src, norm, bufB, dinv,
                                           b2, u2, abf, nullptr, N, 1);
  // --- layer 3: 256 -> 128 ---
  k_gemm_wmma<256, 128><<<RT, 32, 0, stream>>>(abf, wswz3, bufB);
  k_gather_post<128><<<GW, T, 0, stream>>>(rowptr, eidx, src, norm, bufB, dinv,
                                           b3, u3, abf, nullptr, N, 1);
  // --- layer 4: 128 -> 2 ---
  k_gemm_d2<<<cdiv_ll(N, T), T, 0, stream>>>(abf, W4, bufB, N);
  k_gather_small<<<cdiv_ll(N, T), T, 0, stream>>>(rowptr, eidx, src, norm, bufB, dinv,
                                                  b4, bufA, N, 2);
  // --- layer 5: 2 -> 1, final relu straight into d_out ---
  k_gemm_d1<<<cdiv_ll(N, T), T, 0, stream>>>(bufA, W5, bufB, N);
  k_gather_small<<<cdiv_ll(N, T), T, 0, stream>>>(rowptr, eidx, src, norm, bufB, dinv,
                                                  b5, (float*)d_out, N, 1);
}
